// GAT_69509750718999
// MI455X (gfx1250) — matrix operand order repaired and checked by reading, not compiled, not run on hardware
//
#include <hip/hip_runtime.h>
#include <hip/hip_bf16.h>
#include <cmath>

typedef __attribute__((ext_vector_type(16))) _Float16 v16h;
typedef __attribute__((ext_vector_type(8)))  float    v8f;

#define NEG_SLOPE 0.2f

// ---------------------------------------------------------------- utilities

__device__ __forceinline__ void atomicMaxF(float* addr, float val) {
  unsigned int* u = reinterpret_cast<unsigned int*>(addr);
  unsigned int old = *u;
  while (__uint_as_float(old) < val) {
    unsigned int assumed = old;
    old = atomicCAS(u, assumed, __float_as_uint(val));
    if (old == assumed) break;
  }
}

__global__ void k_fill(float* __restrict__ p, float v, long n) {
  long i = (long)blockIdx.x * blockDim.x + threadIdx.x;
  if (i < n) p[i] = v;
}

__global__ void k_init_bias(float* __restrict__ out, const float* __restrict__ b, int N, int C) {
  long i = (long)blockIdx.x * blockDim.x + threadIdx.x;
  if (i < (long)N * C) out[i] = b[i % C];
}

// ------------------------------------------------- operand swizzle kernels
// A-fragment layout (K=128 fixed): for row-tile rt, lane, k-step ks, j in [0,16):
//   out[((rt*32 + lane)*4 + ks)*16 + j] =
//     A[rt*16 + (lane&15)][ks*32 + (lane>>4)*8 + (j<8 ? j : j+8)]
// giving each lane one contiguous 32B block per k-step (2x global_load_b128).

__global__ void k_swzA_f32(const float* __restrict__ in, _Float16* __restrict__ out, long nh) {
  long i = (long)blockIdx.x * blockDim.x + threadIdx.x;
  if (i >= nh) return;
  int  j  = (int)(i & 15);
  long t  = i >> 4;
  int  ks = (int)(t & 3);  t >>= 2;
  int  ln = (int)(t & 31);
  long rt = t >> 5;
  long row = rt * 16 + (ln & 15);
  int  k   = ks * 32 + (ln >> 4) * 8 + (j < 8 ? j : j + 8);
  out[i] = (_Float16)in[row * 128 + k];
}

// Same but applies ELU first (layer-2 GEMM input).
__global__ void k_swzA_elu(const float* __restrict__ in, _Float16* __restrict__ out, long nh) {
  long i = (long)blockIdx.x * blockDim.x + threadIdx.x;
  if (i >= nh) return;
  int  j  = (int)(i & 15);
  long t  = i >> 4;
  int  ks = (int)(t & 3);  t >>= 2;
  int  ln = (int)(t & 31);
  long rt = t >> 5;
  long row = rt * 16 + (ln & 15);
  int  k   = ks * 32 + (ln >> 4) * 8 + (j < 8 ? j : j + 8);
  float v = in[row * 128 + k];
  v = v > 0.0f ? v : (__expf(v) - 1.0f);
  out[i] = (_Float16)v;
}

// B-fragment layout: for k-tile kt, col-tile ct, lane, j in [0,16):
//   out[((kt*(Nc/16) + ct)*32 + lane)*16 + j] =
//     B[kt*32 + (lane>>4)*8 + (j<8 ? j : j+8)][ct*16 + (lane&15)]
__global__ void k_swzB_f32(const float* __restrict__ in, _Float16* __restrict__ out,
                           int K, int Nc) {
  long n = (long)K * Nc;
  long i = (long)blockIdx.x * blockDim.x + threadIdx.x;
  if (i >= n) return;
  int  j  = (int)(i & 15);
  long t  = i >> 4;
  int  ln = (int)(t & 31);  t >>= 5;
  int  nct = Nc / 16;
  int  ct = (int)(t % nct);
  int  kt = (int)(t / nct);
  int  k   = kt * 32 + (ln >> 4) * 8 + (j < 8 ? j : j + 8);
  int  col = ct * 16 + (ln & 15);
  out[i] = (_Float16)in[(size_t)k * Nc + col];
}

// ---------------------------------------------------------------- WMMA GEMM
// C[M,Nc] = A[M,128] x B[128,Nc], operands pre-swizzled to fragment layouts.
// One wave per 16x16 tile: blockIdx.x = row tile, threadIdx.y = col tile.
// C layout per ISA 7.12.2: lanes 0-15 col=l rows 0..7 in v0..7; lanes 16-31 rows 8..15.
template <int K>
__global__ void k_gemm_wmma_f16(const _Float16* __restrict__ Asw,
                                const _Float16* __restrict__ Bsw,
                                float* __restrict__ C, int Nc) {
  constexpr int KS = K / 32;
  const int lane = threadIdx.x & 31;
  const int ct   = threadIdx.y;
  const int nct  = Nc / 16;
  const long rt  = blockIdx.x;

  const v16h* ap = (const v16h*)Asw + ((size_t)rt * 32 + lane) * KS;
  const v16h* bp = (const v16h*)Bsw;

  v8f acc = {};
#pragma unroll
  for (int ks = 0; ks < KS; ++ks) {
    v16h a = ap[ks];
    v16h b = bp[((size_t)ks * nct + ct) * 32 + lane];
    acc = __builtin_amdgcn_wmma_f32_16x16x32_f16(false, a, false, b,
                                                 (short)0, acc, false, false);
  }
  const int half16 = lane >> 4;
  const int l      = lane & 15;
  float* crow = C + ((size_t)rt * 16 + half16 * 8) * Nc + ct * 16 + l;
#pragma unroll
  for (int r = 0; r < 8; ++r) crow[(size_t)r * Nc] = acc[r];
}

// ------------------------------------------------------- per-node attention
// alpha_s[n,h] = dot(h[n, h*32 : h*32+32], a_src[h]); likewise alpha_d.
// One wave per node, wave32 shuffle reduction (C == 32 always here).
__global__ void k_node_alpha(const float* __restrict__ h,
                             const float* __restrict__ a_src,
                             const float* __restrict__ a_dst,
                             float* __restrict__ as, float* __restrict__ ad,
                             int N, int H) {
  int node = blockIdx.x * (blockDim.x >> 5) + (threadIdx.x >> 5);
  int lane = threadIdx.x & 31;
  if (node >= N) return;
  for (int hh = 0; hh < H; ++hh) {
    float v = h[(size_t)node * H * 32 + hh * 32 + lane];
    float s = v * a_src[hh * 32 + lane];
    float d = v * a_dst[hh * 32 + lane];
#pragma unroll
    for (int off = 16; off > 0; off >>= 1) {
      s += __shfl_xor(s, off, 32);
      d += __shfl_xor(d, off, 32);
    }
    if (lane == 0) {
      as[(size_t)node * H + hh] = s;
      ad[(size_t)node * H + hh] = d;
    }
  }
}

// ------------------------------------------------------------- edge kernels
// Edges e < E come from edge_index; e in [E, E+N) are self loops (src=dst=e-E).

template <int H>
__global__ void k_edge_max(const int* __restrict__ src, const int* __restrict__ dst,
                           int E, int N, const float* __restrict__ as,
                           const float* __restrict__ ad, float* __restrict__ mx) {
  long e = (long)blockIdx.x * blockDim.x + threadIdx.x;
  if (e >= (long)E + N) return;
  int s = (e < E) ? src[e] : (int)(e - E);
  int d = (e < E) ? dst[e] : (int)(e - E);
#pragma unroll
  for (int h = 0; h < H; ++h) {
    float lg = as[(size_t)s * H + h] + ad[(size_t)d * H + h];
    lg = lg > 0.0f ? lg : NEG_SLOPE * lg;
    atomicMaxF(&mx[(size_t)d * H + h], lg);
  }
}

template <int H>
__global__ void k_edge_expsum(const int* __restrict__ src, const int* __restrict__ dst,
                              int E, int N, const float* __restrict__ as,
                              const float* __restrict__ ad, const float* __restrict__ mx,
                              float* __restrict__ den) {
  long e = (long)blockIdx.x * blockDim.x + threadIdx.x;
  if (e >= (long)E + N) return;
  int s = (e < E) ? src[e] : (int)(e - E);
  int d = (e < E) ? dst[e] : (int)(e - E);
#pragma unroll
  for (int h = 0; h < H; ++h) {
    float lg = as[(size_t)s * H + h] + ad[(size_t)d * H + h];
    lg = lg > 0.0f ? lg : NEG_SLOPE * lg;
    atomicAdd(&den[(size_t)d * H + h], __expf(lg - mx[(size_t)d * H + h]));
  }
}

// One wave per edge; lane handles HC/32 consecutive channels (head = chan/C).
template <int H, int C>
__global__ void k_edge_message(const int* __restrict__ src, const int* __restrict__ dst,
                               int E, int N, const float* __restrict__ h,
                               const float* __restrict__ as, const float* __restrict__ ad,
                               const float* __restrict__ mx, const float* __restrict__ den,
                               float* __restrict__ out) {
  constexpr int HC  = H * C;
  constexpr int CPL = HC / 32;  // channels per lane
  long wid = (long)blockIdx.x * (blockDim.x >> 5) + (threadIdx.x >> 5);
  if (wid >= (long)E + N) return;
  int lane = threadIdx.x & 31;
  int s = (wid < E) ? src[wid] : (int)(wid - E);
  int d = (wid < E) ? dst[wid] : (int)(wid - E);
  int c0 = lane * CPL;
  int hh = c0 / C;
  float lg = as[(size_t)s * H + hh] + ad[(size_t)d * H + hh];
  lg = lg > 0.0f ? lg : NEG_SLOPE * lg;
  float alpha = __expf(lg - mx[(size_t)d * H + hh]) / den[(size_t)d * H + hh];
#pragma unroll
  for (int j = 0; j < CPL; ++j) {
    atomicAdd(&out[(size_t)d * HC + c0 + j], h[(size_t)s * HC + c0 + j] * alpha);
  }
}

// ---------------------------------------------------------------- launcher

extern "C" void kernel_launch(void* const* d_in, const int* in_sizes, int n_in,
                              void* d_out, int out_size, void* d_ws, size_t ws_size,
                              hipStream_t stream) {
  (void)n_in; (void)out_size; (void)ws_size;

  const float* x    = (const float*)d_in[0];
  const int*   ei   = (const int*)d_in[1];
  const float* W1   = (const float*)d_in[2];
  const float* aS1  = (const float*)d_in[3];
  const float* aD1  = (const float*)d_in[4];
  const float* b1   = (const float*)d_in[5];
  const float* W2   = (const float*)d_in[6];
  const float* aS2  = (const float*)d_in[7];
  const float* aD2  = (const float*)d_in[8];
  const float* b2   = (const float*)d_in[9];

  const int IN = 128, HC1 = 128, C2 = 32;
  const int N = in_sizes[0] / IN;
  const int E = in_sizes[1] / 2;
  const int* src = ei;
  const int* dst = ei + E;

  // workspace carve-up (256B aligned)
  char* w = (char*)d_ws;
  auto carve = [&](size_t bytes) {
    char* p = w;
    w += (bytes + 255) & ~(size_t)255;
    return p;
  };
  _Float16* xh  = (_Float16*)carve((size_t)N * HC1 * sizeof(_Float16)); // A-swz x, later h1act
  _Float16* w1h = (_Float16*)carve((size_t)IN * HC1 * sizeof(_Float16)); // B-swz W1
  _Float16* w2h = (_Float16*)carve((size_t)HC1 * C2 * sizeof(_Float16)); // B-swz W2
  float* h1   = (float*)carve((size_t)N * HC1 * sizeof(float));
  float* out1 = (float*)carve((size_t)N * HC1 * sizeof(float)); // later aliased as h2
  float* as   = (float*)carve((size_t)N * 4 * sizeof(float));   // reused by layer 2
  float* ad   = (float*)carve((size_t)N * 4 * sizeof(float));
  float* mx   = (float*)carve((size_t)N * 4 * sizeof(float));
  float* dn   = (float*)carve((size_t)N * 4 * sizeof(float));

  const dim3 B256(256);
  const long nx  = (long)N * HC1;
  const long ET  = (long)E + N;
  const float NEG_INF = -__builtin_huge_valf();

  // ---- precision conversion + operand swizzle
  k_swzA_f32<<<(nx + 255) / 256, B256, 0, stream>>>(x, xh, nx);
  k_swzB_f32<<<(IN * HC1 + 255) / 256, B256, 0, stream>>>(W1, w1h, IN, HC1);
  k_swzB_f32<<<(HC1 * C2 + 255) / 256, B256, 0, stream>>>(W2, w2h, HC1, C2);

  // ---- layer 1: h1 = x @ W1  (WMMA, 8 col-tiles per block = 8 waves)
  k_gemm_wmma_f16<128><<<dim3(N / 16), dim3(32, HC1 / 16), 0, stream>>>(xh, w1h, h1, HC1);

  k_node_alpha<<<(N + 7) / 8, B256, 0, stream>>>(h1, aS1, aD1, as, ad, N, 4);

  k_fill<<<((long)N * 4 + 255) / 256, B256, 0, stream>>>(mx, NEG_INF, (long)N * 4);
  k_fill<<<((long)N * 4 + 255) / 256, B256, 0, stream>>>(dn, 0.0f, (long)N * 4);
  k_init_bias<<<(nx + 255) / 256, B256, 0, stream>>>(out1, b1, N, HC1); // seg-sum lands on bias

  k_edge_max<4><<<(ET + 255) / 256, B256, 0, stream>>>(src, dst, E, N, as, ad, mx);
  k_edge_expsum<4><<<(ET + 255) / 256, B256, 0, stream>>>(src, dst, E, N, as, ad, mx, dn);
  k_edge_message<4, 32><<<(ET + 7) / 8, B256, 0, stream>>>(src, dst, E, N, h1, as, ad, mx, dn, out1);

  // ---- ELU + convert + A-swizzle for layer-2 GEMM input (reuse xh)
  k_swzA_elu<<<(nx + 255) / 256, B256, 0, stream>>>(out1, xh, nx);

  // ---- layer 2: h2 = elu(h1_out) @ W2  (alias out1, free after swzA_elu)
  float* h2 = out1;
  k_gemm_wmma_f16<128><<<dim3(N / 16), dim3(32, C2 / 16), 0, stream>>>(xh, w2h, h2, C2);

  k_node_alpha<<<(N + 7) / 8, B256, 0, stream>>>(h2, aS2, aD2, as, ad, N, 1);

  k_fill<<<((long)N + 255) / 256, B256, 0, stream>>>(mx, NEG_INF, (long)N);
  k_fill<<<((long)N + 255) / 256, B256, 0, stream>>>(dn, 0.0f, (long)N);
  k_init_bias<<<((long)N * C2 + 255) / 256, B256, 0, stream>>>((float*)d_out, b2, N, C2);

  k_edge_max<1><<<(ET + 255) / 256, B256, 0, stream>>>(src, dst, E, N, as, ad, mx);
  k_edge_expsum<1><<<(ET + 255) / 256, B256, 0, stream>>>(src, dst, E, N, as, ad, mx, dn);
  k_edge_message<1, 32><<<(ET + 7) / 8, B256, 0, stream>>>(src, dst, E, N, h2, as, ad, mx, dn,
                                                           (float*)d_out);
}
